// CM_41162966565199
// MI455X (gfx1250) — matrix operand order repaired and checked by reading, not compiled, not run on hardware
//
#include <hip/hip_runtime.h>

// Problem constants (from reference): N=8, C=64, H=W=128, KS=3
#define NB    8
#define CC    64
#define HH    128
#define WWD   128
#define KTOT  576          // C*9  (GEMM K)
#define RS    136          // padded row stride (elements): data at [4..131], pads at 3 and 132
#define CIS   (4*RS)       // 544 elements per channel (4 staged rows -> 2 output rows)
#define WBS   584          // LDS row stride (bf16 elems) for weight slab: 1168B, 16B-aligned
#define OUTS  17           // padded out-tile stride (floats)
#define OUTROW (128*OUTS)  // 2176 floats per output-row tile
#define RROWS 8            // output rows per workgroup (processed in pairs)

// LDS layout (bytes):
//   wb_s  bf16 [144][584]   = 168192  @ 0        (K-permuted weight slab, this channel group)
//   gt_s  bf16 [64][4][136] = 69632   @ 168192   (zero-padded gt rows h-1..h+2, all 64 in-ch)
//   gr_s  f32  [16][4][136] = 34816   @ 237824   (replicate-padded gr rows, group's channels)
//   out_s f32  [2][128][17] = 17408   @ 272640
//   bias_s f32 [144]        = 576     @ 290048
//   total                   = 290624  (CDNA5 WGP LDS: 320KB)
#define SMEM_BYTES 290624

typedef __attribute__((ext_vector_type(16))) __bf16 v16bf;
typedef __attribute__((ext_vector_type(4)))  __bf16 v4bf;
typedef __attribute__((ext_vector_type(8)))  float  v8f;

// GEMM K is permuted: k = tap*64 + ci, tap = kh*3 + kw. All K steps are constant LDS strides.
__device__ constexpr int klocal0(int s) { return (s & 7) + ((s & 8) << 1); }          // A half -> k (laneLo)
__device__ constexpr int taboff(int tap) { return (tap / 3) * RS + 3 + (tap % 3); }   // row/col offset of tap

__global__ __launch_bounds__(256, 1)
void CM_dynconv_fused(const float* __restrict__ gr, const float* __restrict__ gt,
                      const float* __restrict__ Wc, const float* __restrict__ bc,
                      float* __restrict__ out)
{
    extern __shared__ char smem[];
    __bf16* wb_s   = (__bf16*)(smem);
    __bf16* gt_s   = (__bf16*)(smem + 168192);
    float*  gr_s   = (float*) (smem + 237824);
    float*  out_s  = (float*) (smem + 272640);
    float*  bias_s = (float*) (smem + 290048);

    const int tid    = threadIdx.x;
    const int lane   = tid & 31;
    const int wave   = tid >> 5;            // 8 wave32s
    const int laneHi = (lane >> 4) & 1;
    const int ln16   = lane & 15;

    // 512 blocks: (g, n, hblk)
    const int bid   = blockIdx.x;
    const int g     = bid & 3;              // 16-out-channel group (144 filt channels)
    const int n     = (bid >> 2) & 7;
    const int hbase = (bid >> 5) * RROWS;

    // ---- Stage K-permuted weight slab: wb_s[o][k] = Wc[g*144+o][ (k&63)*9 + (k>>6) ] ----
    for (int idx = tid; idx < 144 * 144; idx += 256) {
        int o   = idx / 144;
        int k   = (idx - o * 144) << 2;     // dest k, multiple of 4 (stays in one 64-block)
        int ci  = k & 63;
        int tap = k >> 6;
        const float* src = Wc + (g * 144 + o) * KTOT + ci * 9 + tap;
        v4bf t;
        t.x = (__bf16)src[0];  t.y = (__bf16)src[9];
        t.z = (__bf16)src[18]; t.w = (__bf16)src[27];
        *(v4bf*)&wb_s[o * WBS + k] = t;
    }

    // ---- Zero static gt pads once; stage bias table ----
    {
        int ci = tid >> 2, q = tid & 3;     // exactly 256 = 64*4
        gt_s[ci * CIS + q * RS + 3]   = (__bf16)0.0f;
        gt_s[ci * CIS + q * RS + 132] = (__bf16)0.0f;
    }
    if (tid < 144) bias_s[tid] = bc[g * 144 + tid];

    const int m = (wave << 4) + ln16;       // this lane's A-matrix row (pixel w)
    // Single-base A pointers (all per-(kc,s) offsets are 16-bit immediates)
    const __bf16* gta0 = gt_s + m + (laneHi ? 8 * CIS : 0);  // output row 0
    const __bf16* gta1 = gta0 + RS;                          // output row 1
    // Three B base registers. Opaque deltas prevent the compiler from folding them back
    // into one base + in-loop v_adds (which cost WMMA->VALU hazard NOPs).
    const __bf16* wbLo = wb_s + ln16 * WBS + (laneHi << 3);  // nt 0..3
    int offMid = 4 * 16 * WBS;  asm("" : "+v"(offMid));
    int offHi  = 8 * 16 * WBS;  asm("" : "+v"(offHi));
    const __bf16* wbMid = (const __bf16*)__builtin_assume_aligned(wbLo + offMid, 16); // nt 4..7
    const __bf16* wbHi  = (const __bf16*)__builtin_assume_aligned(wbLo + offHi, 16);  // nt 8

    for (int rr2 = 0; rr2 < RROWS; rr2 += 2) {
        const int h = hbase + rr2;          // pair of output rows h, h+1
        if (rr2 > 0) __syncthreads();       // previous pair fully consumed before restage

        // ---- Stage gt rows h-1..h+2 (f32 -> packed bf16, vectorized, zero when OOB) ----
        #pragma unroll
        for (int q = 0; q < 4; ++q) {
            const int hs = h + q - 1;
            if (hs >= 0 && hs < HH) {       // uniform branch
                for (int idx = tid; idx < 2048; idx += 256) {   // 64 ci * 32 float4
                    int ci = idx >> 5, k4 = idx & 31;
                    float4 v = ((const float4*)gt)[(((n * CC + ci) * HH + hs) * WWD >> 2) + k4];
                    v4bf t;
                    t.x = (__bf16)v.x; t.y = (__bf16)v.y; t.z = (__bf16)v.z; t.w = (__bf16)v.w;
                    *(v4bf*)&gt_s[ci * CIS + q * RS + 4 + k4 * 4] = t;
                }
            } else {
                v4bf z;
                z.x = (__bf16)0.0f; z.y = (__bf16)0.0f; z.z = (__bf16)0.0f; z.w = (__bf16)0.0f;
                for (int idx = tid; idx < 2048; idx += 256) {
                    int ci = idx >> 5, k4 = idx & 31;
                    *(v4bf*)&gt_s[ci * CIS + q * RS + 4 + k4 * 4] = z;
                }
            }
        }
        // ---- Stage gr rows (f32, vectorized interior + replicate edges) ----
        #pragma unroll
        for (int q = 0; q < 4; ++q) {
            const int hr = min(max(h + q - 1, 0), HH - 1);
            for (int idx = tid; idx < 512; idx += 256) {        // 16 ch * 32 float4
                int cl = idx >> 5, k4 = idx & 31;
                float4 v = ((const float4*)gr)[(((n * CC + g * 16 + cl) * HH + hr) * WWD >> 2) + k4];
                *(float4*)&gr_s[cl * CIS + q * RS + 4 + k4 * 4] = v;
            }
        }
        if (tid < 128) {                    // 16 ch * 4 q * 2 sides
            int cl   = tid >> 3;
            int q    = (tid >> 1) & 3;
            int side = tid & 1;
            int hr   = min(max(h + q - 1, 0), HH - 1);
            const float* row = gr + ((n * CC + g * 16 + cl) * HH + hr) * WWD;
            gr_s[cl * CIS + q * RS + (side ? 132 : 3)] = side ? row[127] : row[0];
        }
        // ---- Zero out tiles ----
        for (int idx = tid; idx < 2 * OUTROW; idx += 256)
            out_s[idx] = 0.0f;

        // ---- Prefetch next pair's fresh rows under this pair's GEMM ----
        if (rr2 + 2 < RROWS) {
            if (h + 3 < HH)
                __builtin_prefetch(gt + (((size_t)(n * CC) + (tid >> 2)) * HH + h + 3) * WWD
                                      + (tid & 3) * 32, 0, 1);
            if (h + 4 < HH)
                __builtin_prefetch(gt + (((size_t)(n * CC) + (tid >> 2)) * HH + h + 4) * WWD
                                      + (tid & 3) * 32, 0, 1);
            if (tid < 64 && h + 3 < HH)
                __builtin_prefetch(gr + (((size_t)(n * CC) + g * 16 + (tid >> 2)) * HH + h + 3) * WWD
                                      + (tid & 3) * 32, 0, 1);
        }
        __syncthreads();

        // ---- Implicit GEMM: [2x128 px] x [K=576] x [144 filt ch], barrier-free ----
        v8f acc0[9], acc1[9];
        #pragma unroll
        for (int nt = 0; nt < 9; ++nt) {
            float b = bias_s[nt * 16 + ln16];
            v8f t;
            #pragma unroll
            for (int i = 0; i < 8; ++i) t[i] = b;
            acc0[nt] = t;
            acc1[nt] = t;
        }

        #pragma unroll
        for (int kc = 0; kc < 18; ++kc) {
            // A fragments, single base + compile-time immediates (K-permuted layout)
            v16bf a0, a1;
            #pragma unroll
            for (int s = 0; s < 16; ++s) {
                const int imm = (((kc & 1) << 5) + klocal0(s)) * CIS + taboff(kc >> 1);
                a0[s] = gta0[imm];
                a1[s] = gta1[imm];
            }
            // 9 B tiles, each feeding 2 WMMAs; all addresses are base + 16-bit immediate
            #pragma unroll
            for (int nt = 0; nt < 9; ++nt) {
                const __bf16* wbn = (nt < 4) ? wbLo : ((nt < 8) ? wbMid : wbHi);
                const int    ntl  = (nt < 4) ? nt   : ((nt < 8) ? nt - 4 : 0);
                v16bf bfr;
                #pragma unroll
                for (int s = 0; s < 16; ++s)
                    bfr[s] = wbn[ntl * 16 * WBS + (kc << 5) + klocal0(s)];
                acc0[nt] = __builtin_amdgcn_wmma_f32_16x16x32_bf16(
                    false, a0, false, bfr, (short)0, acc0[nt], false, false);
                acc1[nt] = __builtin_amdgcn_wmma_f32_16x16x32_bf16(
                    false, a1, false, bfr, (short)0, acc1[nt], false, false);
            }
        }

        // ---- Stage 2: contract filt tiles with replicate-padded gr patches ----
        // acc elem v of tile nt: pixel = wave*16 + laneHi*8 + v, filt ch = nt*16 + (lane&15)
        #pragma unroll
        for (int nt = 0; nt < 9; ++nt) {
            int fl = nt * 16 + ln16;        // 0..143 within group
            int cl = fl / 9;                // local out channel 0..15
            int t9 = fl - cl * 9;
            int kh = t9 / 3;
            int kw = t9 - kh * 3;
            const float* grrow0 = &gr_s[cl * CIS + kh * RS + 3 + kw];  // out row 0 (slots kh..)
            const float* grrow1 = grrow0 + RS;                         // out row 1
            #pragma unroll
            for (int v = 0; v < 8; ++v) {
                int pix = (wave << 4) + (laneHi << 3) + v;
                atomicAdd(&out_s[pix * OUTS + cl],          acc0[nt][v] * grrow0[pix]);
                atomicAdd(&out_s[OUTROW + pix * OUTS + cl], acc1[nt][v] * grrow1[pix]);
            }
        }
        __syncthreads();

        // ---- Store out[n, g*16+cl, h+rr, w], vectorized along w ----
        for (int idx = tid; idx < 1024; idx += 256) {       // 2 rows * 16 ch * 32 float4
            int rr = idx >> 9;
            int rem = idx & 511;
            int cl = rem >> 5;
            int w4 = (rem & 31) * 4;
            float4 v;
            v.x = out_s[rr * OUTROW + (w4 + 0) * OUTS + cl];
            v.y = out_s[rr * OUTROW + (w4 + 1) * OUTS + cl];
            v.z = out_s[rr * OUTROW + (w4 + 2) * OUTS + cl];
            v.w = out_s[rr * OUTROW + (w4 + 3) * OUTS + cl];
            *(float4*)(out + ((n * CC + g * 16 + cl) * HH + h + rr) * WWD + w4) = v;
        }
    }
}

extern "C" void kernel_launch(void* const* d_in, const int* in_sizes, int n_in,
                              void* d_out, int out_size, void* d_ws, size_t ws_size,
                              hipStream_t stream) {
    (void)in_sizes; (void)n_in; (void)out_size; (void)d_ws; (void)ws_size;
    const float* gr = (const float*)d_in[0];
    const float* gt = (const float*)d_in[1];
    const float* Wc = (const float*)d_in[2];
    const float* bc = (const float*)d_in[3];
    float*       out = (float*)d_out;

    // Allow >64KB dynamic LDS (CDNA5 WGP: up to 320KB per workgroup).
    hipFuncSetAttribute((const void*)CM_dynconv_fused,
                        hipFuncAttributeMaxDynamicSharedMemorySize, SMEM_BYTES);

    dim3 grid(4 * NB * (HH / RROWS));   // 512 workgroups: (group, n, row-block)
    dim3 block(256);                    // 8 wave32s
    CM_dynconv_fused<<<grid, block, SMEM_BYTES, stream>>>(gr, gt, Wc, bc, out);
}
